// GVPModel_20237885899306
// MI455X (gfx1250) — matrix operand order repaired and checked by reading, not compiled, not run on hardware
//
#include <hip/hip_runtime.h>

typedef __attribute__((ext_vector_type(16))) _Float16 v16h;
typedef __attribute__((ext_vector_type(8)))  float    v8f;

#define EPS_F  1e-8f
#define LN_EPS 1e-5f

// ---------------------------------------------------------------------------
// Generic WMMA GEMM:  out = act( A[M,K] * B[K,N] + bias )
// A rows are "virtual": up to 4 concatenated segments (ptr, cols, row-index
// mode 0=row/1=src[row]/2=dst[row], dtype f32/f16).  K (padded) must be a
// multiple of 32; columns beyond the segments are zero.
// B is pre-swizzled into WMMA fragment order (see convw_k):
//   Bsw[((kc*nTileN + nt)*32 + lane)*16 + j]  == B[kc*32 + (lane>=16?16:0)+j]
//                                               [nt*16 + (lane&15)]
// so a b-fragment is one contiguous 32B load per lane.
// A is staged in LDS directly in fragment order, so an a-fragment is one
// contiguous 32B LDS load per lane.  Each wave computes two 16x16 N-tiles
// (A-fragment reuse); a 256-thread block covers 16 rows x 256 cols.
// OUTM: 0 = f32 store, 1 = f16 store, 2 = f32 atomic scatter with scale.
// ACT : 0 = none, 1 = relu.
// ---------------------------------------------------------------------------
struct Seg { const void* p; int cols; int mode; int isf16; };

__device__ __forceinline__ int seg_row(const Seg& sg, int row,
                                       const int* __restrict__ srcIdx,
                                       const int* __restrict__ dstIdx)
{
    if (sg.mode == 1) return srcIdx[row];
    if (sg.mode == 2) return dstIdx[row];
    return row;
}

template<int ACT, int OUTM>
__global__ __launch_bounds__(256)
void gemm_wmma(Seg s0, Seg s1, Seg s2, Seg s3, int nseg,
               int M, int Kpad, const _Float16* __restrict__ Bsw, int N,
               const float* __restrict__ bias,
               const int* __restrict__ srcIdx, const int* __restrict__ dstIdx,
               _Float16* __restrict__ outh, float* __restrict__ outf,
               int ldc, int colOff,
               const int* __restrict__ scIdx, const float* __restrict__ scScale)
{
    __shared__ alignas(32) _Float16 AsubF[512];   // fragment order [lane][j]
    const int t    = threadIdx.x;
    const int lane = t & 31;
    const int wid  = t >> 5;
    const int m0   = blockIdx.x * 16;
    const int nTileN = N >> 4;
    const int nt0  = (blockIdx.y * 8 + wid) * 2;

    Seg segs[4] = { s0, s1, s2, s3 };
    int segStart[4];
    {
        int cum = 0;
        #pragma unroll
        for (int i = 0; i < 4; ++i) { segStart[i] = cum; cum += (i < nseg) ? segs[i].cols : 0; }
    }

    // fill-thread coordinates: row fr, even column pair (fc, fc+1)
    const int fr = t >> 4;
    const int fc = (t & 15) << 1;
    const int flane = fr + ((fc >> 4) << 4);
    const int fj    = fc & 15;
    uint* const fillDst = (uint*)&AsubF[flane * 16 + fj];
    const int frow = m0 + fr;

    v8f acc0 = {}, acc1 = {};
    const int nChunks = Kpad >> 5;

    for (int kc = 0; kc < nChunks; ++kc) {
        __syncthreads();
        const int k0 = kc << 5;
        // does one segment cover this whole 32-wide chunk? (uniform)
        int cover = -1;
        #pragma unroll
        for (int i = 0; i < 4; ++i)
            if (i < nseg && k0 >= segStart[i] && (k0 + 32) <= (segStart[i] + segs[i].cols))
                cover = i;

        _Float16 h0 = (_Float16)0.f, h1 = (_Float16)0.f;
        if (cover >= 0) {
            // fast path: 2 contiguous loads, no per-element search
            const Seg sg = segs[cover];
            if (frow < M) {
                const int ridx = seg_row(sg, frow, srcIdx, dstIdx);
                const size_t base = (size_t)ridx * sg.cols + (k0 - segStart[cover]) + fc;
                if (sg.isf16) {
                    const _Float16* p = (const _Float16*)sg.p;
                    h0 = p[base]; h1 = p[base + 1];
                } else {
                    const float* p = (const float*)sg.p;
                    h0 = (_Float16)p[base]; h1 = (_Float16)p[base + 1];
                }
            }
        } else if (frow < M) {
            // generic path (rare: tail / boundary chunks), branch-light
            #pragma unroll
            for (int i = 0; i < 4; ++i) {
                if (i < nseg) {
                    const int c0 = k0 + fc - segStart[i];
                    if (c0 >= 0 && c0 < segs[i].cols) {
                        const int ridx = seg_row(segs[i], frow, srcIdx, dstIdx);
                        const size_t off = (size_t)ridx * segs[i].cols + c0;
                        h0 = segs[i].isf16 ? ((const _Float16*)segs[i].p)[off]
                                           : (_Float16)((const float*)segs[i].p)[off];
                    }
                    const int c1 = c0 + 1;
                    if (c1 >= 0 && c1 < segs[i].cols) {
                        const int ridx = seg_row(segs[i], frow, srcIdx, dstIdx);
                        const size_t off = (size_t)ridx * segs[i].cols + c1;
                        h1 = segs[i].isf16 ? ((const _Float16*)segs[i].p)[off]
                                           : (_Float16)((const float*)segs[i].p)[off];
                    }
                }
            }
        }
        union { _Float16 h[2]; uint u; } pk;
        pk.h[0] = h0; pk.h[1] = h1;
        *fillDst = pk.u;
        __syncthreads();

        const v16h a  = *reinterpret_cast<const v16h*>(&AsubF[lane * 16]);
        const size_t bb = ((size_t)kc * nTileN + nt0) * 512 + (size_t)lane * 16;
        const v16h b0 = *reinterpret_cast<const v16h*>(&Bsw[bb]);
        const v16h b1 = *reinterpret_cast<const v16h*>(&Bsw[bb + 512]);
        acc0 = __builtin_amdgcn_wmma_f32_16x16x32_f16(false, a, false, b0,
                                                      (short)0, acc0, false, false);
        acc1 = __builtin_amdgcn_wmma_f32_16x16x32_f16(false, a, false, b1,
                                                      (short)0, acc1, false, false);
    }

    #pragma unroll
    for (int tile = 0; tile < 2; ++tile) {
        const v8f acc = tile ? acc1 : acc0;
        const int col = (nt0 + tile) * 16 + (lane & 15);
        const float bv = bias ? bias[col] : 0.f;
        #pragma unroll
        for (int i = 0; i < 8; ++i) {
            const int row = m0 + i + ((lane >> 4) << 3);
            if (row >= M) continue;
            float val = acc[i] + bv;
            if (ACT == 1) val = fmaxf(val, 0.f);
            if (OUTM == 0) {
                outf[(size_t)row * ldc + colOff + col] = val;
            } else if (OUTM == 1) {
                outh[(size_t)row * ldc + colOff + col] = (_Float16)val;
            } else {
                const int si = scIdx[row];
                atomicAdd(&outf[(size_t)si * ldc + col], val * scScale[si]);
            }
        }
    }
}

// ---------------------------------------------------------------------------
// Weight conversion f32 [Kt,N] -> f16 fragment-swizzled [Kp,N] (zero padded).
// ---------------------------------------------------------------------------
__global__ void convw_k(const float* __restrict__ src, _Float16* __restrict__ dst,
                        int Kt, int Kp, int N)
{
    const int idx = blockIdx.x * blockDim.x + threadIdx.x;
    if (idx >= Kp * N) return;
    const int j    = idx & 15;
    const int lane = (idx >> 4) & 31;
    const int knt  = idx >> 9;
    const int nTileN = N >> 4;
    const int nt = knt % nTileN;
    const int kc = knt / nTileN;
    const int k  = (kc << 5) + ((lane >= 16) ? 16 : 0) + j;
    const int n  = (nt << 4) + (lane & 15);
    dst[idx] = (k < Kt) ? (_Float16)src[(size_t)k * N + n] : (_Float16)0.f;
}

// ---------------------------------------------------------------------------
// Vector channel path of a GVP:
//   vh[h,k] = sum_c vin[c,k] * wh[c,h] ; vn[h] = ||vh[h,:]|| (clipped)
//   vo[o,k] = sum_h vh[h,k] * wv[h,o] ; optional sigmoid(||vo||) gate
// One wave per item, 8 items per block.  INMODE: 0 = contiguous [item,C,3],
// 1 = edge concat (v[src] 32ch | ev 1ch | v[dst] 32ch), C must be 65.
// SCATTER: atomicAdd vo*inv_cnt[dst] into dv (segment mean aggregation).
// ---------------------------------------------------------------------------
template<int C, int H, int O, int GATE, int INMODE, int SCATTER>
__global__ __launch_bounds__(256)
void vec_path(int nItems,
              const float* __restrict__ vin, const float* __restrict__ vnodes,
              const float* __restrict__ ev,
              const int* __restrict__ srcIdx, const int* __restrict__ dstIdx,
              const float* __restrict__ wh, const float* __restrict__ wv,
              _Float16* __restrict__ vnOut, float* __restrict__ voOut,
              float* __restrict__ dvOut, const float* __restrict__ invCnt)
{
    __shared__ float svin[8][C][3];
    __shared__ float svh [8][H][3];
    const int lane = threadIdx.x & 31;
    const int wid  = threadIdx.x >> 5;
    const int item = blockIdx.x * 8 + wid;
    const bool valid = item < nItems;
    const int e = valid ? item : 0;

    for (int idx = lane; idx < C * 3; idx += 32) {
        int c = idx / 3, k = idx % 3;
        float v = 0.f;
        if (valid) {
            if (INMODE == 0) {
                v = vin[(size_t)e * (C * 3) + idx];
            } else {
                if (c < 32)       v = vnodes[(size_t)srcIdx[e] * 96 + c * 3 + k];
                else if (c == 32) v = ev[(size_t)e * 3 + k];
                else              v = vnodes[(size_t)dstIdx[e] * 96 + (c - 33) * 3 + k];
            }
        }
        svin[wid][c][k] = v;
    }
    __syncthreads();

    for (int h = lane; h < H; h += 32) {
        float a0 = 0.f, a1 = 0.f, a2 = 0.f;
        for (int c = 0; c < C; ++c) {
            float w = wh[c * H + h];
            a0 += svin[wid][c][0] * w;
            a1 += svin[wid][c][1] * w;
            a2 += svin[wid][c][2] * w;
        }
        svh[wid][h][0] = a0; svh[wid][h][1] = a1; svh[wid][h][2] = a2;
        float n = sqrtf(fmaxf(a0 * a0 + a1 * a1 + a2 * a2, EPS_F));
        if (valid && vnOut) vnOut[(size_t)e * H + h] = (_Float16)n;
    }
    __syncthreads();

    if constexpr (O > 0) {
        for (int o = lane; o < O; o += 32) {
            float b0 = 0.f, b1 = 0.f, b2 = 0.f;
            for (int h = 0; h < H; ++h) {
                float w = wv[h * O + o];
                b0 += svh[wid][h][0] * w;
                b1 += svh[wid][h][1] * w;
                b2 += svh[wid][h][2] * w;
            }
            if constexpr (GATE) {
                float n = sqrtf(fmaxf(b0 * b0 + b1 * b1 + b2 * b2, EPS_F));
                float g = 1.f / (1.f + __expf(-n));
                b0 *= g; b1 *= g; b2 *= g;
            }
            if (valid) {
                if constexpr (SCATTER) {
                    int d = dstIdx[e];
                    float sc = invCnt[d];
                    atomicAdd(&dvOut[(size_t)d * (O * 3) + o * 3 + 0], b0 * sc);
                    atomicAdd(&dvOut[(size_t)d * (O * 3) + o * 3 + 1], b1 * sc);
                    atomicAdd(&dvOut[(size_t)d * (O * 3) + o * 3 + 2], b2 * sc);
                } else {
                    voOut[(size_t)e * (O * 3) + o * 3 + 0] = b0;
                    voOut[(size_t)e * (O * 3) + o * 3 + 1] = b1;
                    voOut[(size_t)e * (O * 3) + o * 3 + 2] = b2;
                }
            }
        }
    }
}

// ---------------------------------------------------------------------------
// Residual + GVP LayerNorm on node features (s: 256, v: 32 channels x 3).
// One block (256 threads) per node. addS/addV may be null.
// ---------------------------------------------------------------------------
__global__ __launch_bounds__(256)
void node_ln(int nNodes, float* __restrict__ s, float* __restrict__ v,
             const float* __restrict__ addS, const float* __restrict__ addV,
             const float* __restrict__ gamma, const float* __restrict__ beta)
{
    __shared__ float red[8];
    __shared__ float vbuf[96];
    __shared__ float shval[2];
    const int n = blockIdx.x;
    const int t = threadIdx.x;
    const size_t so = (size_t)n * 256;

    float x = s[so + t] + (addS ? addS[so + t] : 0.f);
    float sum = x;
    for (int o = 16; o; o >>= 1) sum += __shfl_xor(sum, o);
    if ((t & 31) == 0) red[t >> 5] = sum;
    __syncthreads();
    if (t < 8) {
        float z = red[t];
        for (int o = 4; o; o >>= 1) z += __shfl_xor(z, o);
        if (t == 0) shval[0] = z / 256.f;
    }
    __syncthreads();
    float mu = shval[0];
    float d = x - mu;
    float sq = d * d;
    for (int o = 16; o; o >>= 1) sq += __shfl_xor(sq, o);
    if ((t & 31) == 0) red[t >> 5] = sq;
    __syncthreads();
    if (t < 8) {
        float z = red[t];
        for (int o = 4; o; o >>= 1) z += __shfl_xor(z, o);
        if (t == 0) shval[1] = rsqrtf(z / 256.f + LN_EPS);
    }
    if (t < 96) {
        size_t vo = (size_t)n * 96 + t;
        vbuf[t] = v[vo] + (addV ? addV[vo] : 0.f);
    }
    __syncthreads();
    float rstd = shval[1];
    s[so + t] = d * rstd * gamma[t] + beta[t];

    if (t < 32) {
        float n2 = vbuf[t * 3] * vbuf[t * 3] + vbuf[t * 3 + 1] * vbuf[t * 3 + 1] +
                   vbuf[t * 3 + 2] * vbuf[t * 3 + 2];
        n2 = fmaxf(n2, EPS_F);
        for (int o = 16; o; o >>= 1) n2 += __shfl_xor(n2, o);
        if (t == 0) shval[0] = sqrtf(n2 / 32.f);
    }
    __syncthreads();
    if (t < 96) v[(size_t)n * 96 + t] = vbuf[t] / shval[0];
}

// ---------------------------------------------------------------------------
// Initial node LN (s: 6 dims, v: 3 channels).  One thread per node.
// ---------------------------------------------------------------------------
__global__ __launch_bounds__(256)
void node_embed_ln(int nNodes, const float* __restrict__ ns,
                   const float* __restrict__ nv,
                   const float* __restrict__ g6, const float* __restrict__ b6,
                   float* __restrict__ sln6, float* __restrict__ vln9)
{
    int n = blockIdx.x * blockDim.x + threadIdx.x;
    if (n >= nNodes) return;
    float sv[6];
    float mu = 0.f;
    for (int i = 0; i < 6; ++i) { sv[i] = ns[(size_t)n * 6 + i]; mu += sv[i]; }
    mu /= 6.f;
    float var = 0.f;
    for (int i = 0; i < 6; ++i) { float d = sv[i] - mu; var += d * d; }
    var /= 6.f;
    float rstd = rsqrtf(var + LN_EPS);
    for (int i = 0; i < 6; ++i)
        sln6[(size_t)n * 6 + i] = (sv[i] - mu) * rstd * g6[i] + b6[i];

    float vv[9];
    float tot = 0.f;
    for (int c = 0; c < 3; ++c) {
        float n2 = 0.f;
        for (int k = 0; k < 3; ++k) {
            vv[c * 3 + k] = nv[(size_t)n * 9 + c * 3 + k];
            n2 += vv[c * 3 + k] * vv[c * 3 + k];
        }
        tot += fmaxf(n2, EPS_F);
    }
    float rms = sqrtf(tot / 3.f);
    for (int i = 0; i < 9; ++i) vln9[(size_t)n * 9 + i] = vv[i] / rms;
}

// ---------------------------------------------------------------------------
// Fused edge embedding: LN(32) on edge_s, RMS norm on edge_v (1 channel),
// GVP si=32 vi=1 so=32 vo=1 (no activation).  One wave per edge.
// ---------------------------------------------------------------------------
__global__ __launch_bounds__(256)
void edge_embed(int nE, const float* __restrict__ es0, const float* __restrict__ ev0,
                const float* __restrict__ g, const float* __restrict__ b,
                const float* __restrict__ wh, const float* __restrict__ ws,
                const float* __restrict__ bs, const float* __restrict__ wv,
                float* __restrict__ esOut, float* __restrict__ evOut)
{
    const int lane = threadIdx.x & 31;
    const int wid  = threadIdx.x >> 5;
    const int e = blockIdx.x * 8 + wid;
    if (e >= nE) return;

    float x = es0[(size_t)e * 32 + lane];
    float mu = x;
    for (int o = 16; o; o >>= 1) mu += __shfl_xor(mu, o);
    mu *= (1.f / 32.f);
    float d = x - mu;
    float var = d * d;
    for (int o = 16; o; o >>= 1) var += __shfl_xor(var, o);
    var *= (1.f / 32.f);
    float sl = d * rsqrtf(var + LN_EPS) * g[lane] + b[lane];

    float e0 = ev0[(size_t)e * 3 + 0];
    float e1 = ev0[(size_t)e * 3 + 1];
    float e2 = ev0[(size_t)e * 3 + 2];
    float inv = rsqrtf(fmaxf(e0 * e0 + e1 * e1 + e2 * e2, EPS_F));
    e0 *= inv; e1 *= inv; e2 *= inv;
    float wh00 = wh[0], wv00 = wv[0];
    float vh0 = e0 * wh00, vh1 = e1 * wh00, vh2 = e2 * wh00;
    float vn = sqrtf(fmaxf(vh0 * vh0 + vh1 * vh1 + vh2 * vh2, EPS_F));

    float acc = bs[lane] + vn * ws[32 * 32 + lane];
    for (int k = 0; k < 32; ++k) acc += __shfl(sl, k) * ws[k * 32 + lane];
    esOut[(size_t)e * 32 + lane] = acc;
    if (lane < 3) {
        float v = (lane == 0 ? vh0 : (lane == 1 ? vh1 : vh2)) * wv00;
        evOut[(size_t)e * 3 + lane] = v;
    }
}

// ---------------------------------------------------------------------------
// Final scalar LN + global add-pool.  One block per node.
// ---------------------------------------------------------------------------
__global__ __launch_bounds__(256)
void out_ln_pool(int nNodes, const float* __restrict__ outb,
                 const float* __restrict__ g, const float* __restrict__ b,
                 const int* __restrict__ batch, float* __restrict__ pooled)
{
    __shared__ float red[8];
    __shared__ float shval[2];
    const int n = blockIdx.x;
    const int t = threadIdx.x;
    float x = outb[(size_t)n * 256 + t];
    float sum = x;
    for (int o = 16; o; o >>= 1) sum += __shfl_xor(sum, o);
    if ((t & 31) == 0) red[t >> 5] = sum;
    __syncthreads();
    if (t < 8) {
        float z = red[t];
        for (int o = 4; o; o >>= 1) z += __shfl_xor(z, o);
        if (t == 0) shval[0] = z / 256.f;
    }
    __syncthreads();
    float mu = shval[0];
    float d = x - mu;
    float sq = d * d;
    for (int o = 16; o; o >>= 1) sq += __shfl_xor(sq, o);
    if ((t & 31) == 0) red[t >> 5] = sq;
    __syncthreads();
    if (t < 8) {
        float z = red[t];
        for (int o = 4; o; o >>= 1) z += __shfl_xor(z, o);
        if (t == 0) shval[1] = rsqrtf(z / 256.f + LN_EPS);
    }
    __syncthreads();
    float y = d * shval[1] * g[t] + b[t];
    atomicAdd(&pooled[(size_t)batch[n] * 256 + t], y);
}

// ---------------------------------------------------------------------------
// Readout head: out = relu(pooled @ W1 + b1) @ W2 + b2.  One block per graph.
// ---------------------------------------------------------------------------
__global__ __launch_bounds__(256)
void head_k(const float* __restrict__ pooled,
            const float* __restrict__ w1, const float* __restrict__ b1,
            const float* __restrict__ w2, const float* __restrict__ b2,
            float* __restrict__ out)
{
    __shared__ float p[256];
    __shared__ float h[512];
    const int gph = blockIdx.x;
    const int t = threadIdx.x;
    p[t] = pooled[(size_t)gph * 256 + t];
    __syncthreads();
    for (int j = t; j < 512; j += 256) {
        float a = b1[j];
        for (int k = 0; k < 256; ++k) a += p[k] * w1[k * 512 + j];
        h[j] = fmaxf(a, 0.f);
    }
    __syncthreads();
    float a = b2[t];
    for (int k = 0; k < 512; ++k) a += h[k] * w2[k * 256 + t];
    out[(size_t)gph * 256 + t] = a;
}

// ---------------------------------------------------------------------------
// Misc small kernels
// ---------------------------------------------------------------------------
__global__ void count_k(int nE, const int* __restrict__ dst, float* __restrict__ cnt)
{
    int i = blockIdx.x * blockDim.x + threadIdx.x;
    if (i < nE) atomicAdd(&cnt[dst[i]], 1.f);
}

__global__ void inv_k(int nN, float* __restrict__ cnt)
{
    int n = blockIdx.x * blockDim.x + threadIdx.x;
    if (n < nN) cnt[n] = 1.f / fmaxf(cnt[n], 1.f);
}

// ---------------------------------------------------------------------------
// Host orchestration
// ---------------------------------------------------------------------------
static inline int ceildiv(int a, int b) { return (a + b - 1) / b; }

struct LayerP {
    const float *m_wh[3], *m_ws[3], *m_bs[3], *m_wv[3];
    const float *ff_wh[2], *ff_ws[2], *ff_bs[2], *ff_wv[2];
    const float *ln0_g, *ln0_b, *ln1_g, *ln1_b;
};
struct PP {
    const float *Wvln_g, *Wvln_b, *Wv_wh, *Wv_ws, *Wv_bs, *Wv_wv;
    const float *Weln_g, *Weln_b, *We_wh, *We_ws, *We_bs, *We_wv;
    LayerP L[3];
    const float *Woutln_g, *Woutln_b, *Wout_wh, *Wout_ws, *Wout_bs;
    const float *ln_g, *ln_b, *ro1_w, *ro1_b, *ro2_w, *ro2_b;
};

extern "C" void kernel_launch(void* const* d_in, const int* in_sizes, int n_in,
                              void* d_out, int out_size, void* d_ws, size_t ws_size,
                              hipStream_t stream)
{
    const float* node_s = (const float*)d_in[0];
    const float* node_v = (const float*)d_in[1];
    const float* edge_s = (const float*)d_in[2];
    const float* edge_v = (const float*)d_in[3];
    const int*   eidx   = (const int*)d_in[4];
    const int*   batch  = (const int*)d_in[5];

    const int N = in_sizes[0] / 6;
    const int E = in_sizes[4] / 2;
    const int G = out_size / 256;
    const int* srcI = eidx;
    const int* dstI = eidx + E;

    // --- parameter leaf resolution (supports insertion order & jax-sorted) ---
    auto F = [&](int i) { return (const float*)d_in[i]; };
    PP P;
    bool sortedOrder = (n_in > 8 && in_sizes[8] == 1056); // We.ws under sorted keys
    if (!sortedOrder) {
        P.Wvln_g = F(6);  P.Wvln_b = F(7);
        P.Wv_wh = F(8);   P.Wv_ws = F(9);  P.Wv_bs = F(10); P.Wv_wv = F(11);
        P.Weln_g = F(12); P.Weln_b = F(13);
        P.We_wh = F(14);  P.We_ws = F(15); P.We_bs = F(16); P.We_wv = F(17);
        for (int l = 0; l < 3; ++l) {
            int b0 = 18 + 24 * l;
            for (int j = 0; j < 3; ++j) {
                P.L[l].m_wh[j] = F(b0 + 4 * j + 0);
                P.L[l].m_ws[j] = F(b0 + 4 * j + 1);
                P.L[l].m_bs[j] = F(b0 + 4 * j + 2);
                P.L[l].m_wv[j] = F(b0 + 4 * j + 3);
            }
            for (int j = 0; j < 2; ++j) {
                P.L[l].ff_wh[j] = F(b0 + 12 + 4 * j + 0);
                P.L[l].ff_ws[j] = F(b0 + 12 + 4 * j + 1);
                P.L[l].ff_bs[j] = F(b0 + 12 + 4 * j + 2);
                P.L[l].ff_wv[j] = F(b0 + 12 + 4 * j + 3);
            }
            P.L[l].ln0_g = F(b0 + 20); P.L[l].ln0_b = F(b0 + 21);
            P.L[l].ln1_g = F(b0 + 22); P.L[l].ln1_b = F(b0 + 23);
        }
        P.Woutln_g = F(90); P.Woutln_b = F(91);
        P.Wout_wh = F(92);  P.Wout_ws = F(93); P.Wout_bs = F(94);
        P.ln_g = F(95); P.ln_b = F(96);
        P.ro1_w = F(97); P.ro1_b = F(98);
        P.ro2_w = F(99); P.ro2_b = F(100);
    } else {
        P.We_bs = F(6);  P.We_wh = F(7);  P.We_ws = F(8);  P.We_wv = F(9);
        P.Weln_b = F(10); P.Weln_g = F(11);
        P.Wout_bs = F(12); P.Wout_wh = F(13); P.Wout_ws = F(14);
        P.Woutln_b = F(15); P.Woutln_g = F(16);
        P.Wv_bs = F(17); P.Wv_wh = F(18); P.Wv_ws = F(19); P.Wv_wv = F(20);
        P.Wvln_b = F(21); P.Wvln_g = F(22);
        for (int l = 0; l < 3; ++l) {
            int b0 = 23 + 24 * l;
            for (int j = 0; j < 2; ++j) {
                P.L[l].ff_bs[j] = F(b0 + 4 * j + 0);
                P.L[l].ff_wh[j] = F(b0 + 4 * j + 1);
                P.L[l].ff_ws[j] = F(b0 + 4 * j + 2);
                P.L[l].ff_wv[j] = F(b0 + 4 * j + 3);
            }
            P.L[l].ln0_b = F(b0 + 8);  P.L[l].ln0_g = F(b0 + 9);
            P.L[l].ln1_b = F(b0 + 10); P.L[l].ln1_g = F(b0 + 11);
            for (int j = 0; j < 3; ++j) {
                P.L[l].m_bs[j] = F(b0 + 12 + 4 * j + 0);
                P.L[l].m_wh[j] = F(b0 + 12 + 4 * j + 1);
                P.L[l].m_ws[j] = F(b0 + 12 + 4 * j + 2);
                P.L[l].m_wv[j] = F(b0 + 12 + 4 * j + 3);
            }
        }
        P.ln_b = F(95); P.ln_g = F(96);
        P.ro1_b = F(97); P.ro1_w = F(98);
        P.ro2_b = F(99); P.ro2_w = F(100);
    }

    // --- workspace layout ---
    size_t off = 0;
    auto alloc = [&](size_t bytes) {
        void* p = (char*)d_ws + off;
        off = (off + bytes + 255) & ~(size_t)255;
        return p;
    };
    float* sN   = (float*)alloc((size_t)N * 256 * 4);   // scalar node features
    float* vN   = (float*)alloc((size_t)N * 96 * 4);    // vector node features
    float* sln6 = (float*)alloc((size_t)N * 6 * 4);
    float* vln9 = (float*)alloc((size_t)N * 9 * 4);
    float* esB  = (float*)alloc((size_t)E * 32 * 4);
    float* evB  = (float*)alloc((size_t)E * 3 * 4);
    float* cnt  = (float*)alloc((size_t)N * 4);         // becomes inv_cnt
    float* ds   = (float*)alloc((size_t)N * 256 * 4);
    float* dv   = (float*)alloc((size_t)N * 96 * 4);
    float* fs2  = (float*)alloc((size_t)N * 256 * 4);   // ff1 out / out-stage buf
    float* fvh  = (float*)alloc((size_t)N * 64 * 3 * 4);
    float* fv2  = (float*)alloc((size_t)N * 96 * 4);
    float* pooled = (float*)alloc((size_t)G * 256 * 4);
    _Float16* vn_e  = (_Float16*)alloc((size_t)N * 32 * 2);
    _Float16* vnff  = (_Float16*)alloc((size_t)N * 64 * 2);
    _Float16* vnff1 = (_Float16*)alloc((size_t)N * 64 * 2);
    _Float16* vno   = (_Float16*)alloc((size_t)N * 32 * 2);
    _Float16* fs1   = (_Float16*)alloc((size_t)N * 1024 * 2);
    _Float16* vn0   = (_Float16*)alloc((size_t)E * 65 * 2);
    _Float16* vn1   = (_Float16*)alloc((size_t)E * 32 * 2);
    _Float16* vn2   = (_Float16*)alloc((size_t)E * 32 * 2);
    _Float16* ms1   = (_Float16*)alloc((size_t)E * 256 * 2);
    _Float16* ms2   = (_Float16*)alloc((size_t)E * 256 * 2);
    float* mva = (float*)alloc((size_t)E * 96 * 4);
    float* mvb = (float*)alloc((size_t)E * 96 * 4);
    _Float16* Wembf16  = (_Float16*)alloc((size_t)64 * 256 * 2);
    _Float16* Wm0f16   = (_Float16*)alloc((size_t)640 * 256 * 2);
    _Float16* Wm1f16   = (_Float16*)alloc((size_t)288 * 256 * 2);
    _Float16* Wm2f16   = (_Float16*)alloc((size_t)288 * 256 * 2);
    _Float16* Wff0f16  = (_Float16*)alloc((size_t)320 * 1024 * 2);
    _Float16* Wff1f16  = (_Float16*)alloc((size_t)1088 * 256 * 2);
    _Float16* Woutf16  = (_Float16*)alloc((size_t)288 * 256 * 2);
    (void)ws_size;

    const Seg Z = { nullptr, 0, 0, 0 };
    const int eTiles = ceildiv(E, 16);
    const int nTiles = ceildiv(N, 16);

    // --- in-degree counts -> inverse ---
    hipMemsetAsync(cnt, 0, (size_t)N * 4, stream);
    count_k<<<ceildiv(E, 256), 256, 0, stream>>>(E, dstI, cnt);
    inv_k<<<ceildiv(N, 256), 256, 0, stream>>>(N, cnt);

    // --- node embedding ---
    node_embed_ln<<<ceildiv(N, 256), 256, 0, stream>>>(N, node_s, node_v,
        P.Wvln_g, P.Wvln_b, sln6, vln9);
    vec_path<3, 32, 32, 0, 0, 0><<<ceildiv(N, 8), 256, 0, stream>>>(N, vln9,
        nullptr, nullptr, nullptr, nullptr, P.Wv_wh, P.Wv_wv, vn_e, vN,
        nullptr, nullptr);
    convw_k<<<ceildiv(64 * 256, 256), 256, 0, stream>>>(P.Wv_ws, Wembf16, 38, 64, 256);
    {
        Seg a = { sln6, 6, 0, 0 }, b = { vn_e, 32, 0, 1 };
        gemm_wmma<0, 0><<<dim3(nTiles, 1), 256, 0, stream>>>(a, b, Z, Z, 2,
            N, 64, Wembf16, 256, P.Wv_bs, nullptr, nullptr,
            nullptr, sN, 256, 0, nullptr, nullptr);
    }

    // --- edge embedding (fused LN + GVP) ---
    edge_embed<<<ceildiv(E, 8), 256, 0, stream>>>(E, edge_s, edge_v,
        P.Weln_g, P.Weln_b, P.We_wh, P.We_ws, P.We_bs, P.We_wv, esB, evB);

    // --- GNN layers ---
    for (int l = 0; l < 3; ++l) {
        const LayerP& LP = P.L[l];
        convw_k<<<ceildiv(640 * 256, 256), 256, 0, stream>>>(LP.m_ws[0], Wm0f16, 609, 640, 256);
        convw_k<<<ceildiv(288 * 256, 256), 256, 0, stream>>>(LP.m_ws[1], Wm1f16, 288, 288, 256);
        convw_k<<<ceildiv(288 * 256, 256), 256, 0, stream>>>(LP.m_ws[2], Wm2f16, 288, 288, 256);
        convw_k<<<ceildiv(320 * 1024, 256), 256, 0, stream>>>(LP.ff_ws[0], Wff0f16, 320, 320, 1024);
        convw_k<<<ceildiv(1088 * 256, 256), 256, 0, stream>>>(LP.ff_ws[1], Wff1f16, 1088, 1088, 256);

        // message GVP0 (vector path: edge concat; gated)
        vec_path<65, 65, 32, 1, 1, 0><<<ceildiv(E, 8), 256, 0, stream>>>(E,
            nullptr, vN, evB, srcI, dstI, LP.m_wh[0], LP.m_wv[0], vn0, mva,
            nullptr, nullptr);
        {   // scalar GEMM with fused concat gather [s_j | e_s | s_i | vn0]
            Seg a = { sN, 256, 1, 0 }, b = { esB, 32, 0, 0 };
            Seg c = { sN, 256, 2, 0 }, d = { vn0, 65, 0, 1 };
            gemm_wmma<1, 1><<<dim3(eTiles, 1), 256, 0, stream>>>(a, b, c, d, 4,
                E, 640, Wm0f16, 256, LP.m_bs[0], srcI, dstI,
                ms1, nullptr, 256, 0, nullptr, nullptr);
        }
        // message GVP1
        vec_path<32, 32, 32, 1, 0, 0><<<ceildiv(E, 8), 256, 0, stream>>>(E,
            mva, nullptr, nullptr, nullptr, nullptr, LP.m_wh[1], LP.m_wv[1],
            vn1, mvb, nullptr, nullptr);
        {
            Seg a = { ms1, 256, 0, 1 }, b = { vn1, 32, 0, 1 };
            gemm_wmma<1, 1><<<dim3(eTiles, 1), 256, 0, stream>>>(a, b, Z, Z, 2,
                E, 288, Wm1f16, 256, LP.m_bs[1], nullptr, nullptr,
                ms2, nullptr, 256, 0, nullptr, nullptr);
        }
        // message GVP2 + segment-mean aggregation
        hipMemsetAsync(ds, 0, (size_t)N * 256 * 4, stream);
        hipMemsetAsync(dv, 0, (size_t)N * 96 * 4, stream);
        vec_path<32, 32, 32, 0, 0, 1><<<ceildiv(E, 8), 256, 0, stream>>>(E,
            mvb, nullptr, nullptr, nullptr, dstI, LP.m_wh[2], LP.m_wv[2],
            vn2, nullptr, dv, cnt);
        {
            Seg a = { ms2, 256, 0, 1 }, b = { vn2, 32, 0, 1 };
            gemm_wmma<0, 2><<<dim3(eTiles, 1), 256, 0, stream>>>(a, b, Z, Z, 2,
                E, 288, Wm2f16, 256, LP.m_bs[2], nullptr, nullptr,
                nullptr, ds, 256, 0, dstI, cnt);
        }
        // residual + LN0
        node_ln<<<N, 256, 0, stream>>>(N, sN, vN, ds, dv, LP.ln0_g, LP.ln0_b);

        // feed-forward GVPs
        vec_path<32, 64, 64, 1, 0, 0><<<ceildiv(N, 8), 256, 0, stream>>>(N,
            vN, nullptr, nullptr, nullptr, nullptr, LP.ff_wh[0], LP.ff_wv[0],
            vnff, fvh, nullptr, nullptr);
        {
            Seg a = { sN, 256, 0, 0 }, b = { vnff, 64, 0, 1 };
            gemm_wmma<1, 1><<<dim3(nTiles, 4), 256, 0, stream>>>(a, b, Z, Z, 2,
                N, 320, Wff0f16, 1024, LP.ff_bs[0], nullptr, nullptr,
                fs1, nullptr, 1024, 0, nullptr, nullptr);
        }
        vec_path<64, 64, 32, 0, 0, 0><<<ceildiv(N, 8), 256, 0, stream>>>(N,
            fvh, nullptr, nullptr, nullptr, nullptr, LP.ff_wh[1], LP.ff_wv[1],
            vnff1, fv2, nullptr, nullptr);
        {
            Seg a = { fs1, 1024, 0, 1 }, b = { vnff1, 64, 0, 1 };
            gemm_wmma<0, 0><<<dim3(nTiles, 1), 256, 0, stream>>>(a, b, Z, Z, 2,
                N, 1088, Wff1f16, 256, LP.ff_bs[1], nullptr, nullptr,
                nullptr, fs2, 256, 0, nullptr, nullptr);
        }
        // residual + LN1
        node_ln<<<N, 256, 0, stream>>>(N, sN, vN, fs2, fv2, LP.ln1_g, LP.ln1_b);
    }

    // --- output block ---
    node_ln<<<N, 256, 0, stream>>>(N, sN, vN, nullptr, nullptr,
        P.Woutln_g, P.Woutln_b);
    vec_path<32, 32, 0, 0, 0, 0><<<ceildiv(N, 8), 256, 0, stream>>>(N,
        vN, nullptr, nullptr, nullptr, nullptr, P.Wout_wh, nullptr,
        vno, nullptr, nullptr, nullptr);
    convw_k<<<ceildiv(288 * 256, 256), 256, 0, stream>>>(P.Wout_ws, Woutf16, 288, 288, 256);
    {
        Seg a = { sN, 256, 0, 0 }, b = { vno, 32, 0, 1 };
        gemm_wmma<1, 0><<<dim3(nTiles, 1), 256, 0, stream>>>(a, b, Z, Z, 2,
            N, 288, Woutf16, 256, P.Wout_bs, nullptr, nullptr,
            nullptr, fs2, 256, 0, nullptr, nullptr);
    }
    hipMemsetAsync(pooled, 0, (size_t)G * 256 * 4, stream);
    out_ln_pool<<<N, 256, 0, stream>>>(N, fs2, P.ln_g, P.ln_b, batch, pooled);
    head_k<<<G, 256, 0, stream>>>(pooled, P.ro1_w, P.ro1_b, P.ro2_w, P.ro2_b,
                                  (float*)d_out);
}
